// BatchConv1d_65824668778890
// MI455X (gfx1250) — matrix operand order repaired and checked by reading, not compiled, not run on hardware
//
#include <hip/hip_runtime.h>
#include <hip/hip_bf16.h>
#include <stdint.h>

// ---------------------------------------------------------------------------
// BatchConv1d (dynamic per-query conv kernels) for MI455X / gfx1250.
//   GEMM1: kern[b][t][i][c] = q[b,i,:]·Wk[c*3+t,:] + bk     (25.8 GFLOP)
//   GEMM2: out[b,i,j] = sum_t sum_c kern[b][t][i][c]*k[b,j+t-1,c]
//          + bias[b,i] + bias_b                              (51.5 GFLOP)
// bf16 WMMA (V_WMMA_F32_16X16X32_BF16, fp32 accumulate); k-window staged to
// LDS with GLOBAL_LOAD_ASYNC_TO_LDS_B128 (ASYNCcnt); all staging vectorized
// to 16B transactions.
// ---------------------------------------------------------------------------

typedef __attribute__((ext_vector_type(16))) __bf16 v16bf;
typedef __attribute__((ext_vector_type(8)))  float  v8f;

union FragBF {
    v16bf v;
    uint4 q[2];
};

__device__ __forceinline__ unsigned short f2bf(float f) {
    union { float f; unsigned int u; } cv;
    cv.f = f;
    unsigned int u = cv.u;
    u += 0x7FFFu + ((u >> 16) & 1u);   // round-to-nearest-even
    return (unsigned short)(u >> 16);
}

__device__ __forceinline__ unsigned pack2(float a, float b) {
    return (unsigned)f2bf(a) | ((unsigned)f2bf(b) << 16);
}

__device__ __forceinline__ v16bf load_frag_lds(const unsigned short* p0,
                                               const unsigned short* p1) {
    FragBF f;
    f.q[0] = *(const uint4*)p0;   // ds_load_b128
    f.q[1] = *(const uint4*)p1;   // ds_load_b128
    return f.v;
}

constexpr int Bc = 16, QL = 1024, KL = 1024, QS = 512, KS = 512, KW = 3;
constexpr int OD = KS * KW;  // 1536

// workspace layout (bytes)
constexpr size_t WKB_OFF  = 0;                                        // bf16 [3][512][512]
constexpr size_t BKP_OFF  = WKB_OFF  + (size_t)KW * KS * QS * 2;      // f32  [3][512]
constexpr size_t KB_OFF   = BKP_OFF  + (size_t)KW * KS * 4;           // bf16 [16][1024][512]
constexpr size_t KERN_OFF = KB_OFF   + (size_t)Bc * KL * KS * 2;      // bf16 [16][3][1024][512]
constexpr size_t BIAS_OFF = KERN_OFF + (size_t)Bc * KW * QL * KS * 2; // f32  [16][1024]
constexpr size_t WS_NEED  = BIAS_OFF + (size_t)Bc * QL * 4;           // ~65.6 MB

// --------------------------- prep kernels ----------------------------------

// Permute Wk [o=c*3+t][q] -> WkB[t][c][q] bf16 (4 q-elems/thread), bk -> bkp[t][c].
__global__ __launch_bounds__(256) void prep_wk_kernel(
    const float* __restrict__ Wk, const float* __restrict__ bk,
    unsigned short* __restrict__ WkB, float* __restrict__ bkp) {
    int idx = blockIdx.x * 256 + threadIdx.x;
    if (idx < OD * (QS / 4)) {
        int o = idx / (QS / 4), qi = (idx % (QS / 4)) * 4;
        int c = o / KW, t = o % KW;
        float4 v = *(const float4*)(Wk + (size_t)o * QS + qi);
        uint2 pk;
        pk.x = pack2(v.x, v.y);
        pk.y = pack2(v.z, v.w);
        *(uint2*)(WkB + ((size_t)(t * KS + c)) * QS + qi) = pk;
    } else if (idx < OD * (QS / 4) + OD) {
        int o = idx - OD * (QS / 4);
        bkp[(o % KW) * KS + (o / KW)] = bk[o];
    }
}

// Convert k fp32 -> bf16 (8 elems/thread).
__global__ __launch_bounds__(256) void prep_k_kernel(
    const float* __restrict__ k, unsigned short* __restrict__ kB) {
    size_t base = ((size_t)blockIdx.x * 256 + threadIdx.x) * 8;
    float4 a = *(const float4*)(k + base);
    float4 b = *(const float4*)(k + base + 4);
    uint4 pk;
    pk.x = pack2(a.x, a.y);
    pk.y = pack2(a.z, a.w);
    pk.z = pack2(b.x, b.y);
    pk.w = pack2(b.z, b.w);
    *(uint4*)(kB + base) = pk;
}

// bias[b,i] = q[b,i,:]·Wb + bb   (one wave32 per row, shuffle reduce)
__global__ __launch_bounds__(256) void bias_kernel(
    const float* __restrict__ q, const float* __restrict__ Wb,
    const float* __restrict__ bb, float* __restrict__ bias) {
    int row  = blockIdx.x * 8 + (threadIdx.x >> 5);
    int lane = threadIdx.x & 31;
    const float* qr = q + (size_t)row * QS;
    float s = 0.f;
#pragma unroll
    for (int it = 0; it < QS / 128; ++it) {
        int c = it * 128 + lane * 4;
        float4 qv = *(const float4*)(qr + c);
        float4 wv = *(const float4*)(Wb + c);
        s += qv.x * wv.x + qv.y * wv.y + qv.z * wv.z + qv.w * wv.w;
    }
#pragma unroll
    for (int off = 16; off >= 1; off >>= 1) s += __shfl_xor(s, off, 32);
    if (lane == 0) bias[row] = s + bb[0];
}

// --------------------------- GEMM1 -----------------------------------------
// [16384,512] x [512,1536] tiled 64M x 128N per WG, K-chunks of 32 in LDS.
// 8 waves: wave (wm=w&3, wn=w>>2) owns 16M x 64N (4 wmma accumulators).
__global__ __launch_bounds__(256) void gemm1_kernel(
    const float* __restrict__ q, const unsigned short* __restrict__ WkB,
    const float* __restrict__ bkp, unsigned short* __restrict__ kern) {
    __shared__ unsigned short qs[64 * 40];     // padded stride 40 shorts = 80B
    __shared__ unsigned short wss[128 * 40];
    const int tid  = threadIdx.x;
    const int lane = tid & 31;
    const int w    = tid >> 5;
    const int wm = w & 3, wn = w >> 2;
    const int gm0 = blockIdx.x * 64;       // global row (b*QL + i)
    const int o0  = blockIdx.y * 128;      // flat [t][c] column block
    const int t   = o0 / KS;
    const int c0  = o0 % KS;
    const int b   = gm0 / QL;
    const int i0  = gm0 % QL;

    v8f acc[4] = {};
    const int amrow  = wm * 16 + (lane & 15);
    const int kabase = (lane < 16) ? 0 : 8;    // A-fragment K base (ISA layout)
    const int kbbase = (lane < 16) ? 0 : 8;    // B-fragment K base

    for (int kc = 0; kc < QS / 32; ++kc) {
        __syncthreads();
        // q chunk: 64 rows x 32 ch fp32 -> bf16; 512 float4-groups, 2/thread
#pragma unroll
        for (int ii = 0; ii < 2; ++ii) {
            int g = tid + ii * 256;
            int row = g >> 3, ch = (g & 7) * 4;
            float4 v = *(const float4*)(q + (size_t)(gm0 + row) * QS + kc * 32 + ch);
            uint2 pk;
            pk.x = pack2(v.x, v.y);
            pk.y = pack2(v.z, v.w);
            *(uint2*)(qs + row * 40 + ch) = pk;
        }
        // Wk chunk: 128 c-rows x 32 q-ch bf16; 512 uint4-groups, 2/thread
#pragma unroll
        for (int ii = 0; ii < 2; ++ii) {
            int g = tid + ii * 256;
            int row = g >> 2, ch8 = (g & 3) * 8;
            uint4 v = *(const uint4*)(WkB + ((size_t)(t * KS + c0 + row)) * QS +
                                      kc * 32 + ch8);
            *(uint4*)(wss + row * 40 + ch8) = v;
        }
        __syncthreads();
        const unsigned short* ap = &qs[amrow * 40 + kabase];
        v16bf a = load_frag_lds(ap, ap + 16);
        v16bf bfr[4];
#pragma unroll
        for (int bnt = 0; bnt < 4; ++bnt) {
            int cl = wn * 64 + bnt * 16 + (lane & 15);
            const unsigned short* bp = &wss[cl * 40 + kbbase];
            bfr[bnt] = load_frag_lds(bp, bp + 16);
        }
#pragma unroll
        for (int bnt = 0; bnt < 4; ++bnt)
            acc[bnt] = __builtin_amdgcn_wmma_f32_16x16x32_bf16(
                false, a, false, bfr[bnt], (short)0, acc[bnt], false, false);
    }
    const int mo = (lane & 16) ? 8 : 0;
#pragma unroll
    for (int bnt = 0; bnt < 4; ++bnt) {
        int c = c0 + wn * 64 + bnt * 16 + (lane & 15);
        float bkv = bkp[t * KS + c];
#pragma unroll
        for (int r = 0; r < 8; ++r) {
            int m = wm * 16 + r + mo;
            kern[(((size_t)(b * 3 + t) * QL + (i0 + m)) * KS) + c] =
                f2bf(acc[bnt][r] + bkv);
        }
    }
}

// --------------------------- GEMM2 -----------------------------------------
// Per (b, 64-row M tile, 128-col N tile): banded GEMM over K = (t,c).
// One 130-row k window per K-chunk (covers all 3 shifts), staged with
// GLOBAL_LOAD_ASYNC_TO_LDS_B128; kern A-slices staged with uint4 loads.
__global__ __launch_bounds__(256) void gemm2_kernel(
    const unsigned short* __restrict__ kern, const unsigned short* __restrict__ kB,
    const float* __restrict__ bias, const float* __restrict__ bias_b,
    float* __restrict__ out) {
    __shared__ unsigned short as3[3 * 64 * 40];
    __shared__ unsigned short bs[130 * 40];
    __shared__ float biases[64];
    const int tid  = threadIdx.x;
    const int lane = tid & 31;
    const int w    = tid >> 5;
    const int wm = w & 3, wn = w >> 2;
    const int b  = blockIdx.z;
    const int i0 = blockIdx.x * 64;
    const int n0 = blockIdx.y * 128;

    if (tid < 64) biases[tid] = bias[b * QL + i0 + tid];
    const float bb0 = bias_b[0];

    v8f acc[4] = {};
    const int amrow  = wm * 16 + (lane & 15);
    const int kabase = (lane < 16) ? 0 : 8;
    const int kbbase = (lane < 16) ? 0 : 8;

    for (int kc = 0; kc < KS / 32; ++kc) {
        __syncthreads();
        // A slices (all 3 t): 3*64 rows x 32 ch bf16 = 768 uint4-groups, 3/thread
#pragma unroll
        for (int ii = 0; ii < 3; ++ii) {
            int g = tid + ii * 256;
            int t = g >> 8, rem = g & 255;
            int row = rem >> 2, ch8 = (rem & 3) * 8;
            uint4 v = *(const uint4*)(kern +
                (((size_t)(b * 3 + t) * QL + (i0 + row)) * KS) + kc * 32 + ch8);
            *(uint4*)(as3 + t * (64 * 40) + row * 40 + ch8) = v;
        }
        // k window: rows n0-1 .. n0+128 (130 rows) x 32 ch = 520 uint4-groups.
        // In-range rows: async global->LDS copy (no VGPR round trip).
        // Out-of-range rows (band edges): zero-fill with plain ds stores.
#pragma unroll
        for (int ii = 0; ii < 3; ++ii) {
            int g = tid + ii * 256;
            if (g < 520) {
                int row = g >> 2, ch8 = (g & 3) * 8;
                int j = n0 + row - 1;
                unsigned ldsoff = (unsigned)(size_t)(bs + row * 40 + ch8);
                if (j >= 0 && j < KL) {
                    unsigned long long gaddr = (unsigned long long)(size_t)(
                        kB + ((size_t)b * KL + j) * KS + kc * 32 + ch8);
                    asm volatile("global_load_async_to_lds_b128 %0, %1, off"
                                 :: "v"(ldsoff), "v"(gaddr) : "memory");
                } else {
                    uint4 z = {0u, 0u, 0u, 0u};
                    *(uint4*)(bs + row * 40 + ch8) = z;
                }
            }
        }
        asm volatile("s_wait_asynccnt 0" ::: "memory");
        __syncthreads();
#pragma unroll
        for (int t = 0; t < 3; ++t) {
            const unsigned short* ap = &as3[t * (64 * 40) + amrow * 40 + kabase];
            v16bf a = load_frag_lds(ap, ap + 16);
            v16bf bfr[4];
#pragma unroll
            for (int bnt = 0; bnt < 4; ++bnt) {
                int jl = wn * 64 + bnt * 16 + (lane & 15);
                const unsigned short* bp = &bs[(jl + t) * 40 + kbbase];
                bfr[bnt] = load_frag_lds(bp, bp + 16);
            }
#pragma unroll
            for (int bnt = 0; bnt < 4; ++bnt)
                acc[bnt] = __builtin_amdgcn_wmma_f32_16x16x32_bf16(
                    false, a, false, bfr[bnt], (short)0, acc[bnt], false, false);
        }
    }
    const int mo = (lane & 16) ? 8 : 0;
#pragma unroll
    for (int bnt = 0; bnt < 4; ++bnt) {
        int j = n0 + wn * 64 + bnt * 16 + (lane & 15);
#pragma unroll
        for (int r = 0; r < 8; ++r) {
            int m = wm * 16 + r + mo;
            out[((size_t)b * QL + (i0 + m)) * KL + j] = acc[bnt][r] + biases[m] + bb0;
        }
    }
}

// --------------------------- launch ----------------------------------------

extern "C" void kernel_launch(void* const* d_in, const int* in_sizes, int n_in,
                              void* d_out, int out_size, void* d_ws, size_t ws_size,
                              hipStream_t stream) {
    (void)in_sizes; (void)n_in; (void)out_size;
    if (ws_size < WS_NEED) return;  // workspace too small to run

    const float* q      = (const float*)d_in[0];
    const float* k      = (const float*)d_in[1];
    const float* Wk     = (const float*)d_in[2];
    const float* bk     = (const float*)d_in[3];
    const float* Wb     = (const float*)d_in[4];
    const float* bb     = (const float*)d_in[5];
    const float* bias_b = (const float*)d_in[6];
    float* out = (float*)d_out;

    char* ws = (char*)d_ws;
    unsigned short* WkB  = (unsigned short*)(ws + WKB_OFF);
    float*          bkp  = (float*)(ws + BKP_OFF);
    unsigned short* kB   = (unsigned short*)(ws + KB_OFF);
    unsigned short* kern = (unsigned short*)(ws + KERN_OFF);
    float*          bias = (float*)(ws + BIAS_OFF);

    prep_wk_kernel<<<(OD * (QS / 4) + OD + 255) / 256, 256, 0, stream>>>(Wk, bk, WkB, bkp);
    prep_k_kernel<<<(int)((size_t)Bc * KL * KS / 8 / 256), 256, 0, stream>>>(k, kB);
    bias_kernel<<<Bc * QL / 8, 256, 0, stream>>>(q, Wb, bb, bias);

    gemm1_kernel<<<dim3(Bc * QL / 64, OD / 128), 256, 0, stream>>>(q, WkB, bkp, kern);
    gemm2_kernel<<<dim3(QL / 64, KL / 128, Bc), 256, 0, stream>>>(kern, kB, bias, bias_b, out);
}
